// HybridGNNTransformer_68264210203158
// MI455X (gfx1250) — matrix operand order repaired and checked by reading, not compiled, not run on hardware
//
#include <hip/hip_runtime.h>
#include <hip/hip_bf16.h>
#include <math.h>

typedef __attribute__((ext_vector_type(16))) _Float16 v16h;
typedef __attribute__((ext_vector_type(8)))  _Float16 v8h;
typedef __attribute__((ext_vector_type(8)))  float    v8f;

#define B_SZ   128
#define NROIS  116
#define NBANDS 4
#define FIN    72
#define HID    64
#define HEADS  4
#define HD     16
#define SEQ    464            // NROIS*NBANDS
#define M2     (B_SZ*SEQ)     // 59392 rows

// ---------------------------------------------------------------- helpers

__device__ __forceinline__ float lrelu(float z) { return z >= 0.f ? z : 0.2f * z; }

__device__ __forceinline__ v8f zero8() {
  v8f z;
#pragma unroll
  for (int i = 0; i < 8; ++i) z[i] = 0.0f;
  return z;
}

// A fragment: 16x32 f16. Per lane: 16 contiguous halves (K kb..kb+15)
// -> two 16B vector loads.
__device__ __forceinline__ v16h load_a_frag(const _Float16* __restrict__ A, int lda, int m0, int k0) {
  const int lane = threadIdx.x & 31;
  const int m  = m0 + (lane & 15);
  const int kb = k0 + ((lane & 16) ? 16 : 0);
  const _Float16* p = A + (size_t)m * lda + kb;
  v8h lo = *(const v8h*)p;          // K kb .. kb+7
  v8h hi = *(const v8h*)(p + 8);    // K kb+8 .. kb+15
  return __builtin_shufflevector(lo, hi, 0, 1, 2, 3, 4, 5, 6, 7,
                                          8, 9, 10, 11, 12, 13, 14, 15);
}

// B fragment from K-major storage Bt[n][k] (i.e. B(k,n) = Bt[n*ldt+k]).
// Per lane: 16 contiguous halves -> two 16B vector loads.
__device__ __forceinline__ v16h load_bt_frag(const _Float16* __restrict__ Bt, int ldt, int n0, int k0) {
  const int lane = threadIdx.x & 31;
  const int n  = n0 + (lane & 15);
  const int kb = k0 + ((lane & 16) ? 16 : 0);
  const _Float16* p = Bt + (size_t)n * ldt + kb;
  v8h lo = *(const v8h*)p;
  v8h hi = *(const v8h*)(p + 8);
  return __builtin_shufflevector(lo, hi, 0, 1, 2, 3, 4, 5, 6, 7,
                                          8, 9, 10, 11, 12, 13, 14, 15);
}

// B fragment from f32 weights stored [N][K] row-major (B(k,n) = W[n][k]).
// Contiguous float4 global loads + convert.
__device__ __forceinline__ v16h load_bT_frag_f32(const float* __restrict__ W, int ldw, int k0, int n0) {
  const int lane = threadIdx.x & 31;
  const int n  = n0 + (lane & 15);
  const int kb = k0 + ((lane & 16) ? 16 : 0);
  const float4* p = (const float4*)(W + (size_t)n * ldw + kb);
  float4 w0 = p[0], w1 = p[1], w2 = p[2], w3 = p[3];
  v16h f;
  f[0]  = (_Float16)w0.x; f[1]  = (_Float16)w0.y; f[2]  = (_Float16)w0.z; f[3]  = (_Float16)w0.w;
  f[4]  = (_Float16)w1.x; f[5]  = (_Float16)w1.y; f[6]  = (_Float16)w1.z; f[7]  = (_Float16)w1.w;
  f[8]  = (_Float16)w2.x; f[9]  = (_Float16)w2.y; f[10] = (_Float16)w2.z; f[11] = (_Float16)w2.w;
  f[12] = (_Float16)w3.x; f[13] = (_Float16)w3.y; f[14] = (_Float16)w3.z; f[15] = (_Float16)w3.w;
  return f;
}

__device__ __forceinline__ v8f wmma32(v16h a, v16h b, v8f c) {
  return __builtin_amdgcn_wmma_f32_16x16x32_f16(false, a, false, b, (short)0, c, false, false);
}

// ---------------------------------------------------------------- stage 1: per-(batch,band) GAT

__global__ void __launch_bounds__(256) stage1_gat(
    const float* __restrict__ nf, const float* __restrict__ adj,
    const float* __restrict__ Wg, const float* __restrict__ aSrc,
    const float* __restrict__ aDst, const float* __restrict__ We,
    const float* __restrict__ aE, const float* __restrict__ bias,
    float* __restrict__ x32)
{
  const int tid = threadIdx.x, lane = tid & 31, wave = tid >> 5;
  const int b = blockIdx.x >> 2, band = blockIdx.x & 3;

  __shared__ __align__(16) char smem[37632];
  _Float16* xh  = (_Float16*)smem;            // [128][96] x (f16, zero-padded)
  _Float16* wlT = (_Float16*)(smem + 24576);  // [64][96]  W^T, K-major
  _Float16* aT  = (_Float16*)smem;            // [128][128] alpha^T (reuses xh/wlT)
  __shared__ __align__(16) _Float16 hhT[64][128];  // h^T (col-major h), K-major B operand
  __shared__ float sS[4][128], dS[4][128];
  __shared__ float cH[4], loopw[128], sumE[128], loopE[128];

  // stage x (f32 -> f16, zero-pad rows>=116, k>=72)
  for (int idx = tid; idx < 128 * 96; idx += 256) {
    int r = idx / 96, k = idx - r * 96;
    float v = 0.f;
    if (r < NROIS && k < FIN) v = nf[((size_t)b * SEQ + band * NROIS + r) * FIN + k];
    xh[idx] = (_Float16)v;
  }
  // stage W transposed: wlT[n][k] = W[k][n]
  for (int idx = tid; idx < 64 * 96; idx += 256) {
    int n = idx / 96, k = idx - n * 96;
    float v = (k < FIN) ? Wg[((size_t)band * FIN + k) * HID + n] : 0.f;
    wlT[idx] = (_Float16)v;
  }
  __syncthreads();

  const int mo = (lane & 16) ? 8 : 0, nn = lane & 15;

  // h = x @ W : wave w owns m-tile w, all 4 n-tiles, K = 96 (3 steps)
  {
    v8f acc[4];
#pragma unroll
    for (int nt = 0; nt < 4; ++nt) acc[nt] = zero8();
#pragma unroll
    for (int kt = 0; kt < 3; ++kt) {
      v16h a = load_a_frag(xh, 96, wave * 16, kt * 32);
#pragma unroll
      for (int nt = 0; nt < 4; ++nt)
        acc[nt] = wmma32(a, load_bt_frag(wlT, 96, nt * 16, kt * 32), acc[nt]);
    }
#pragma unroll
    for (int nt = 0; nt < 4; ++nt)
#pragma unroll
      for (int r = 0; r < 8; ++r)
        hhT[nt * 16 + nn][wave * 16 + r + mo] = (_Float16)acc[nt][r];
  }
  __syncthreads();

  // per-node attention terms s, d ; per-head edge coeff c ; self-loop weights
  for (int p = tid; p < HEADS * NROIS; p += 256) {
    int hd = p / NROIS, i = p - hd * NROIS;
    float ss = 0.f, dd2 = 0.f;
#pragma unroll
    for (int dd = 0; dd < HD; ++dd) {
      float h = (float)hhT[hd * 16 + dd][i];
      ss  += h * aSrc[band * 64 + hd * 16 + dd];
      dd2 += h * aDst[band * 64 + hd * 16 + dd];
    }
    sS[hd][i] = ss; dS[hd][i] = dd2;
  }
  if (tid < 4) {
    float c = 0.f;
#pragma unroll
    for (int dd = 0; dd < HD; ++dd)
      c += We[band * 64 + tid * 16 + dd] * aE[band * 64 + tid * 16 + dd];
    cH[tid] = c;
  }
  const float* adjB = adj + ((size_t)b * NBANDS + band) * NROIS * NROIS;
  if (tid < NROIS) {
    int j = tid; int cnt = 0; float ssum = 0.f;
    for (int i = 0; i < NROIS; ++i) {
      float a = adjB[i * NROIS + j];
      if (a != 0.f) { ++cnt; ssum += a; }
    }
    loopw[j] = cnt > 0 ? ssum / (float)cnt : 0.f;
  }
  __syncthreads();

  // per-head: build alpha^T (unnormalized exp) then out = alpha^T @ h via WMMA
  for (int hd = 0; hd < HEADS; ++hd) {
    if (tid < 128) {
      int j = tid;
      if (j < NROIS) {
        float dj = dS[hd][j], c = cH[hd];
        float lgl = lrelu(sS[hd][j] + dj + loopw[j] * c);
        float mx = lgl;
        for (int i = 0; i < NROIS; ++i) {
          float a = adjB[i * NROIS + j];
          if (a != 0.f) mx = fmaxf(mx, lrelu(sS[hd][i] + dj + a * c));
        }
        float sum = 0.f;
        for (int i = 0; i < 128; ++i) {
          float e = 0.f;
          if (i < NROIS) {
            float a = adjB[i * NROIS + j];
            if (a != 0.f) e = __expf(lrelu(sS[hd][i] + dj + a * c) - mx);
          }
          aT[j * 128 + i] = (_Float16)e;
          sum += e;
        }
        float el = __expf(lgl - mx);
        sumE[j] = sum + el; loopE[j] = el;
      } else {
        for (int i = 0; i < 128; ++i) aT[j * 128 + i] = (_Float16)0.f;
        sumE[j] = 1.f; loopE[j] = 0.f;
      }
    }
    __syncthreads();
    v8f acc = zero8();
#pragma unroll
    for (int kt = 0; kt < 4; ++kt)
      acc = wmma32(load_a_frag(aT, 128, wave * 16, kt * 32),
                   load_bt_frag(&hhT[0][0], 128, hd * 16, kt * 32), acc);
#pragma unroll
    for (int r = 0; r < 8; ++r) {
      int m = wave * 16 + r + mo;
      if (m < NROIS) {
        int col = hd * 16 + nn;
        float v = (acc[r] + loopE[m] * (float)hhT[col][m]) / sumE[m] + bias[band * HID + col];
        x32[((size_t)b * SEQ + m * NBANDS + band) * HID + col] = v;  // seq = roi*4+band
      }
    }
    __syncthreads();
  }
}

// ---------------------------------------------------------------- row LayerNorm (+residual, + optional 2nd LN)

__global__ void __launch_bounds__(256) ln_rows(
    const float* __restrict__ xin, const float* __restrict__ yin,
    const float* __restrict__ g1, const float* __restrict__ b1,
    const float* __restrict__ g2, const float* __restrict__ b2,
    float* __restrict__ xout, _Float16* __restrict__ x16out, int rows)
{
  int row = blockIdx.x * blockDim.x + threadIdx.x;
  if (row >= rows) return;
  float v[HID];
  float mu = 0.f;
#pragma unroll
  for (int c = 0; c < HID; ++c) {
    float t = xin[(size_t)row * HID + c];
    if (yin) t += yin[(size_t)row * HID + c];
    v[c] = t; mu += t;
  }
  mu *= (1.f / HID);
  float var = 0.f;
#pragma unroll
  for (int c = 0; c < HID; ++c) { float d = v[c] - mu; var += d * d; }
  var *= (1.f / HID);
  float inv = 1.f / sqrtf(var + 1e-5f);
#pragma unroll
  for (int c = 0; c < HID; ++c) v[c] = (v[c] - mu) * inv * g1[c] + b1[c];
  if (g2) {
    float mu2 = 0.f;
#pragma unroll
    for (int c = 0; c < HID; ++c) mu2 += v[c];
    mu2 *= (1.f / HID);
    float var2 = 0.f;
#pragma unroll
    for (int c = 0; c < HID; ++c) { float d = v[c] - mu2; var2 += d * d; }
    var2 *= (1.f / HID);
    float inv2 = 1.f / sqrtf(var2 + 1e-5f);
#pragma unroll
    for (int c = 0; c < HID; ++c) v[c] = (v[c] - mu2) * inv2 * g2[c] + b2[c];
  }
#pragma unroll
  for (int c = 0; c < HID; ++c) {
    xout[(size_t)row * HID + c] = v[c];
    if (x16out) x16out[(size_t)row * HID + c] = (_Float16)v[c];
  }
}

// ---------------------------------------------------------------- generic GEMM: Y[M,N] = X[M,K] @ W^T + b

template <int NT>
__global__ void __launch_bounds__(256) gemm_wmma(
    const _Float16* __restrict__ A, int lda, const float* __restrict__ W,
    const float* __restrict__ bias, int K,
    float* __restrict__ out32, _Float16* __restrict__ out16, int relu)
{
  __shared__ __align__(16) _Float16 Al[128][32];
  __shared__ __align__(16) _Float16 Bt[NT * 16][32];   // K-major weight tile
  const int tid = threadIdx.x, lane = tid & 31, wave = tid >> 5;
  const int m0g = blockIdx.x * 128;
  const int N = NT * 16;

  v8f acc[NT];
#pragma unroll
  for (int nt = 0; nt < NT; ++nt) acc[nt] = zero8();

  for (int kc = 0; kc < K; kc += 32) {
    __syncthreads();
    // prefetch next K-chunk (global_prefetch_b8)
    if (kc + 32 < K) {
      __builtin_prefetch(A + (size_t)(m0g + (tid >> 1)) * lda + kc + 32, 0, 1);
      __builtin_prefetch(W + (size_t)(tid % N) * K + kc + 32, 0, 1);
    }
    // A chunk: 128 rows x 64B, 16B vector copies
    for (int idx = tid; idx < 512; idx += 256) {
      int r = idx >> 2, part = idx & 3;
      ((uint4*)&Al[r][0])[part] =
          ((const uint4*)(A + (size_t)(m0g + r) * lda + kc))[part];
    }
    // W chunk -> Bt[n][k] (row-contiguous global reads, f32->f16)
    for (int idx = tid; idx < N * 8; idx += 256) {
      int n = idx >> 3, c = (idx & 7) * 4;
      const float* wp = W + (size_t)n * K + kc + c;
      _Float16* dp = &Bt[n][c];
      dp[0] = (_Float16)wp[0]; dp[1] = (_Float16)wp[1];
      dp[2] = (_Float16)wp[2]; dp[3] = (_Float16)wp[3];
    }
    __syncthreads();
    v16h a = load_a_frag(&Al[0][0], 32, wave * 16, 0);
#pragma unroll
    for (int nt = 0; nt < NT; ++nt)
      acc[nt] = wmma32(a, load_bt_frag(&Bt[0][0], 32, nt * 16, 0), acc[nt]);
  }
  const int mo = (lane & 16) ? 8 : 0;
#pragma unroll
  for (int nt = 0; nt < NT; ++nt) {
    int col = nt * 16 + (lane & 15);
    float bv = bias ? bias[col] : 0.f;
#pragma unroll
    for (int r = 0; r < 8; ++r) {
      size_t row = (size_t)m0g + wave * 16 + r + mo;
      float v = acc[nt][r] + bv;
      if (relu) v = v > 0.f ? v : 0.f;
      if (out32) out32[row * N + col] = v;
      if (out16) out16[row * N + col] = (_Float16)v;
    }
  }
}

// ---------------------------------------------------------------- flash attention (seq=464, HD=16), wave32 shuffles

__global__ void __launch_bounds__(256) attn_flash(
    const _Float16* __restrict__ qkv, _Float16* __restrict__ out16)
{
  const int tid = threadIdx.x, lane = tid & 31, wave = tid >> 5;
  int blk = blockIdx.x;
  const int qb = blk & 3; blk >>= 2;
  const int h  = blk & 3; const int b = blk >> 2;
  const int qbase = qb * 128;

  __shared__ __align__(16) _Float16 Qt[128][32];  // A operand (d padded to 32)
  __shared__ __align__(16) _Float16 Kc[64][32];   // Bt for S: Kc[kv][d]
  __shared__ __align__(16) _Float16 VT[16][64];   // Bt for O: VT[d][kv]
  __shared__ __align__(16) _Float16 Pb[128][64];  // A operand for P@V

  for (int idx = tid; idx < 128 * 32; idx += 256) {
    int r = idx >> 5, d = idx & 31;
    int grow = qbase + r;
    float v = 0.f;
    if (d < HD && grow < SEQ)
      v = 0.25f * (float)qkv[((size_t)b * SEQ + grow) * 192 + h * 16 + d];  // 1/sqrt(HD)
    Qt[r][d] = (_Float16)v;
  }
  __syncthreads();

  v16h aq = load_a_frag(&Qt[0][0], 32, wave * 16, 0);
  v8f accO = zero8();
  float rm[8], rl[8];
#pragma unroll
  for (int r = 0; r < 8; ++r) { rm[r] = -1e30f; rl[r] = 0.f; }
  const int mo = (lane & 16) ? 8 : 0, nn = lane & 15;

  for (int kv0 = 0; kv0 < SEQ; kv0 += 64) {
    int vkv = SEQ - kv0; if (vkv > 64) vkv = 64;
    __syncthreads();
    if (kv0 + 64 < SEQ)
      __builtin_prefetch(qkv + ((size_t)b * SEQ + kv0 + 64 + (tid & 63)) * 192 + 64, 0, 1);
    // K chunk: Kc[kv][d] (row-contiguous global reads)
    for (int idx = tid; idx < 64 * 32; idx += 256) {
      int c = idx >> 5, d = idx & 31;
      float v = 0.f;
      if (d < HD && c < vkv) v = (float)qkv[((size_t)b * SEQ + kv0 + c) * 192 + 64 + h * 16 + d];
      Kc[c][d] = (_Float16)v;
    }
    // V chunk transposed: VT[d][kv]
    for (int idx = tid; idx < 64 * 16; idx += 256) {
      int r = idx >> 4, d = idx & 15;
      float v = 0.f;
      if (r < vkv) v = (float)qkv[((size_t)b * SEQ + kv0 + r) * 192 + 128 + h * 16 + d];
      VT[d][r] = (_Float16)v;
    }
    __syncthreads();

    v8f s[4];
#pragma unroll
    for (int nt = 0; nt < 4; ++nt) {
      v8f z = zero8();
      s[nt] = wmma32(aq, load_bt_frag(&Kc[0][0], 32, nt * 16, 0), z);
    }

    // online softmax in registers (one row = one 16-lane group of the C frag)
#pragma unroll
    for (int r = 0; r < 8; ++r) {
      float mx = -1e30f;
#pragma unroll
      for (int nt = 0; nt < 4; ++nt) {
        int col = nt * 16 + nn;
        float v = (col < vkv) ? s[nt][r] : -1e30f;
        mx = fmaxf(mx, v);
      }
      mx = fmaxf(mx, __shfl_xor(mx, 1)); mx = fmaxf(mx, __shfl_xor(mx, 2));
      mx = fmaxf(mx, __shfl_xor(mx, 4)); mx = fmaxf(mx, __shfl_xor(mx, 8));
      float mnew = fmaxf(rm[r], mx);
      float scale = __expf(rm[r] - mnew);
      float ssum = 0.f;
      int prow = wave * 16 + r + mo;
#pragma unroll
      for (int nt = 0; nt < 4; ++nt) {
        int col = nt * 16 + nn;
        float e = (col < vkv) ? __expf(s[nt][r] - mnew) : 0.f;
        Pb[prow][col] = (_Float16)e;
        ssum += e;
      }
      ssum += __shfl_xor(ssum, 1); ssum += __shfl_xor(ssum, 2);
      ssum += __shfl_xor(ssum, 4); ssum += __shfl_xor(ssum, 8);
      rl[r] = rl[r] * scale + ssum;
      rm[r] = mnew;
      accO[r] *= scale;
    }
    __syncthreads();
    accO = wmma32(load_a_frag(&Pb[0][0], 64, wave * 16, 0),  load_bt_frag(&VT[0][0], 64, 0, 0),  accO);
    accO = wmma32(load_a_frag(&Pb[0][0], 64, wave * 16, 32), load_bt_frag(&VT[0][0], 64, 0, 32), accO);
  }

#pragma unroll
  for (int r = 0; r < 8; ++r) {
    int grow = qbase + wave * 16 + r + mo;
    if (grow < SEQ) {
      float v = accO[r] / (rl[r] + 1e-20f);
      out16[((size_t)b * SEQ + grow) * HID + h * 16 + nn] = (_Float16)v;
    }
  }
}

// ---------------------------------------------------------------- stage 3: inter-band GAT + fusion attention + LNs

__global__ void __launch_bounds__(256) stage3_fuse(
    const _Float16* __restrict__ x16, const float* __restrict__ interW,
    const float* __restrict__ inAs, const float* __restrict__ inAd,
    const float* __restrict__ inWe, const float* __restrict__ inAe,
    const float* __restrict__ inBias,
    const float* __restrict__ fQkvW, const float* __restrict__ fQkvB,
    const float* __restrict__ fOutW, const float* __restrict__ fOutB,
    const float* __restrict__ fusg, const float* __restrict__ fusb,
    const float* __restrict__ ilng, const float* __restrict__ ilnb,
    float* __restrict__ out)
{
  const int tid = threadIdx.x, lane = tid & 31, wave = tid >> 5;
  const size_t tok0 = (size_t)blockIdx.x * 64;   // 64 band-tokens = 16 ROIs

  __shared__ __align__(16) _Float16 w3[64][64];   // interW^T (K-major)
  __shared__ __align__(16) _Float16 hh3[64][64];
  __shared__ __align__(16) _Float16 io[64][64];
  __shared__ __align__(16) _Float16 Kb[64][64];
  __shared__ __align__(16) _Float16 Vb[64][64];
  __shared__ __align__(16) _Float16 qm[16][64];
  __shared__ __align__(16) _Float16 qh[16][64];
  __shared__ __align__(16) _Float16 oh[16][64];
  __shared__ float fz[16][64];
  __shared__ float s3[4][64], d3[4][64], c3[4];

  const _Float16* Ax = x16 + tok0 * HID;
  const int mo = (lane & 16) ? 8 : 0, nn = lane & 15;

  // stage interW transposed (w3[n][k] = interW[k][n])
  for (int idx = tid; idx < 64 * 64; idx += 256) {
    int k = idx >> 6, n = idx & 63;
    w3[n][k] = (_Float16)interW[(size_t)k * 64 + n];
  }
  __syncthreads();

  // h = x @ interW   (M=64)
  {
    int mt = wave & 3, ntA = wave >> 2;
    v8f a0 = zero8(), a1 = zero8();
#pragma unroll
    for (int kt = 0; kt < 2; ++kt) {
      v16h a = load_a_frag(Ax, 64, mt * 16, kt * 32);
      a0 = wmma32(a, load_bt_frag(&w3[0][0], 64, ntA * 16, kt * 32), a0);
      a1 = wmma32(a, load_bt_frag(&w3[0][0], 64, (ntA + 2) * 16, kt * 32), a1);
    }
#pragma unroll
    for (int r = 0; r < 8; ++r) {
      hh3[mt * 16 + r + mo][ntA * 16 + nn]       = (_Float16)a0[r];
      hh3[mt * 16 + r + mo][(ntA + 2) * 16 + nn] = (_Float16)a1[r];
    }
  }
  __syncthreads();

  { // s, d per (head, token); c per head
    int hd = tid >> 6, t = tid & 63;
    float ss = 0.f, dd2 = 0.f;
#pragma unroll
    for (int dd = 0; dd < HD; ++dd) {
      float h = (float)hh3[t][hd * 16 + dd];
      ss += h * inAs[hd * 16 + dd]; dd2 += h * inAd[hd * 16 + dd];
    }
    s3[hd][t] = ss; d3[hd][t] = dd2;
  }
  if (tid < 4) {
    float c = 0.f;
#pragma unroll
    for (int dd = 0; dd < HD; ++dd) c += inWe[tid * 16 + dd] * inAe[tid * 16 + dd];
    c3[tid] = c;
  }
  __syncthreads();

  // inter GAT (A = ones, loopw = 1): tiny -> VALU
  if (tid < 64) {
    int j = tid, q = j >> 2;
#pragma unroll
    for (int hd = 0; hd < HEADS; ++hd) {
      float c = c3[hd], dj = d3[hd][j];
      float lgl = lrelu(s3[hd][j] + dj + c);
      float lg[4], mx = lgl;
#pragma unroll
      for (int i = 0; i < 4; ++i) { lg[i] = lrelu(s3[hd][q * 4 + i] + dj + c); mx = fmaxf(mx, lg[i]); }
      float e[4], el = __expf(lgl - mx), sum = el;
#pragma unroll
      for (int i = 0; i < 4; ++i) { e[i] = __expf(lg[i] - mx); sum += e[i]; }
#pragma unroll
      for (int dd = 0; dd < HD; ++dd) {
        int col = hd * 16 + dd;
        float o = el * (float)hh3[j][col];
#pragma unroll
        for (int i = 0; i < 4; ++i) o += e[i] * (float)hh3[q * 4 + i][col];
        io[j][col] = (_Float16)(o / sum + inBias[col]);
      }
    }
  }
  __syncthreads();

  // mean query per ROI
  for (int idx = tid; idx < 16 * 64; idx += 256) {
    int q = idx >> 6, col = idx & 63;
    float v = 0.25f * ((float)io[q * 4][col] + (float)io[q * 4 + 1][col] +
                       (float)io[q * 4 + 2][col] + (float)io[q * 4 + 3][col]);
    qm[q][col] = (_Float16)v;
  }
  __syncthreads();

  // K,V projections (M=64), weights straight from global f32 (contiguous b128)
  {
    int mt = wave & 3, ntA = wave >> 2;
    for (int pv = 0; pv < 2; ++pv) {
      const float* Wp = fQkvW + (size_t)(64 + pv * 64) * 64;
      const float* bp = fQkvB + 64 + pv * 64;
      v8f a0 = zero8(), a1 = zero8();
#pragma unroll
      for (int kt = 0; kt < 2; ++kt) {
        v16h a = load_a_frag(&io[0][0], 64, mt * 16, kt * 32);
        a0 = wmma32(a, load_bT_frag_f32(Wp, 64, kt * 32, ntA * 16), a0);
        a1 = wmma32(a, load_bT_frag_f32(Wp, 64, kt * 32, (ntA + 2) * 16), a1);
      }
#pragma unroll
      for (int r = 0; r < 8; ++r) {
        int m = mt * 16 + r + mo;
        if (pv == 0) {
          Kb[m][ntA * 16 + nn]       = (_Float16)(a0[r] + bp[ntA * 16 + nn]);
          Kb[m][(ntA + 2) * 16 + nn] = (_Float16)(a1[r] + bp[(ntA + 2) * 16 + nn]);
        } else {
          Vb[m][ntA * 16 + nn]       = (_Float16)(a0[r] + bp[ntA * 16 + nn]);
          Vb[m][(ntA + 2) * 16 + nn] = (_Float16)(a1[r] + bp[(ntA + 2) * 16 + nn]);
        }
      }
    }
  }
  // q projection (M=16): waves 0..3
  if (wave < 4) {
    v8f acc = zero8();
#pragma unroll
    for (int kt = 0; kt < 2; ++kt)
      acc = wmma32(load_a_frag(&qm[0][0], 64, 0, kt * 32),
                   load_bT_frag_f32(fQkvW, 64, kt * 32, wave * 16), acc);
#pragma unroll
    for (int r = 0; r < 8; ++r)
      qh[r + mo][wave * 16 + nn] = (_Float16)(acc[r] + fQkvB[wave * 16 + nn]);
  }
  __syncthreads();

  // fusion attention over 4 band tokens per ROI
  if (tid < 64) {
    int q = tid >> 2, hd = tid & 3;
    float sc[4], mx = -1e30f;
#pragma unroll
    for (int i = 0; i < 4; ++i) {
      float s = 0.f;
#pragma unroll
      for (int dd = 0; dd < HD; ++dd)
        s += (float)qh[q][hd * 16 + dd] * (float)Kb[q * 4 + i][hd * 16 + dd];
      sc[i] = 0.25f * s; mx = fmaxf(mx, sc[i]);
    }
    float e[4], sum = 0.f;
#pragma unroll
    for (int i = 0; i < 4; ++i) { e[i] = __expf(sc[i] - mx); sum += e[i]; }
#pragma unroll
    for (int dd = 0; dd < HD; ++dd) {
      float o = 0.f;
#pragma unroll
      for (int i = 0; i < 4; ++i) o += e[i] * (float)Vb[q * 4 + i][hd * 16 + dd];
      oh[q][hd * 16 + dd] = (_Float16)(o / sum);
    }
  }
  __syncthreads();

  // output projection (M=16)
  if (wave < 4) {
    v8f acc = zero8();
#pragma unroll
    for (int kt = 0; kt < 2; ++kt)
      acc = wmma32(load_a_frag(&oh[0][0], 64, 0, kt * 32),
                   load_bT_frag_f32(fOutW, 64, kt * 32, wave * 16), acc);
#pragma unroll
    for (int r = 0; r < 8; ++r)
      fz[r + mo][wave * 16 + nn] = acc[r] + fOutB[wave * 16 + nn];
  }
  __syncthreads();

  // fusion LN then final inter LN, write output row per ROI
  if (tid < 16) {
    int q = tid;
    float mu = 0.f;
    for (int c = 0; c < HID; ++c) mu += fz[q][c];
    mu *= (1.f / HID);
    float var = 0.f;
    for (int c = 0; c < HID; ++c) { float d = fz[q][c] - mu; var += d * d; }
    var *= (1.f / HID);
    float inv = 1.f / sqrtf(var + 1e-5f);
    float mu2 = 0.f;
    for (int c = 0; c < HID; ++c) mu2 += (fz[q][c] - mu) * inv * fusg[c] + fusb[c];
    mu2 *= (1.f / HID);
    float var2 = 0.f;
    for (int c = 0; c < HID; ++c) {
      float y = (fz[q][c] - mu) * inv * fusg[c] + fusb[c];
      float d = y - mu2; var2 += d * d;
    }
    var2 *= (1.f / HID);
    float inv2 = 1.f / sqrtf(var2 + 1e-5f);
    size_t orow = (size_t)blockIdx.x * 16 + q;
    for (int c = 0; c < HID; ++c) {
      float y = (fz[q][c] - mu) * inv * fusg[c] + fusb[c];
      out[orow * HID + c] = (y - mu2) * inv2 * ilng[c] + ilnb[c];
    }
  }
}

// ---------------------------------------------------------------- launcher

extern "C" void kernel_launch(void* const* d_in, const int* in_sizes, int n_in,
                              void* d_out, int out_size, void* d_ws, size_t ws_size,
                              hipStream_t stream) {
  (void)in_sizes; (void)n_in; (void)out_size; (void)ws_size;

  const float* nf     = (const float*)d_in[0];
  const float* adj    = (const float*)d_in[1];
  const float* iW     = (const float*)d_in[2];
  const float* iAs    = (const float*)d_in[3];
  const float* iAd    = (const float*)d_in[4];
  const float* iWe    = (const float*)d_in[5];
  const float* iAe    = (const float*)d_in[6];
  const float* iBias  = (const float*)d_in[7];
  const float* iLng   = (const float*)d_in[8];
  const float* iLnb   = (const float*)d_in[9];
  const float* tLng   = (const float*)d_in[10];
  const float* tLnb   = (const float*)d_in[11];
  const float* qkvW   = (const float*)d_in[12];
  const float* qkvB   = (const float*)d_in[13];
  const float* outW   = (const float*)d_in[14];
  const float* outB   = (const float*)d_in[15];
  const float* ff1W   = (const float*)d_in[16];
  const float* ff1B   = (const float*)d_in[17];
  const float* ff2W   = (const float*)d_in[18];
  const float* ff2B   = (const float*)d_in[19];
  const float* ln1g   = (const float*)d_in[20];
  const float* ln1b   = (const float*)d_in[21];
  const float* ln2g   = (const float*)d_in[22];
  const float* ln2b   = (const float*)d_in[23];
  const float* interW = (const float*)d_in[24];
  const float* inAs   = (const float*)d_in[25];
  const float* inAd   = (const float*)d_in[26];
  const float* inWe   = (const float*)d_in[27];
  const float* inAe   = (const float*)d_in[28];
  const float* inBias = (const float*)d_in[29];
  const float* fQkvW  = (const float*)d_in[30];
  const float* fQkvB  = (const float*)d_in[31];
  const float* fOutW  = (const float*)d_in[32];
  const float* fOutB  = (const float*)d_in[33];
  const float* fusg   = (const float*)d_in[34];
  const float* fusb   = (const float*)d_in[35];
  const float* ilng   = (const float*)d_in[36];
  const float* ilnb   = (const float*)d_in[37];

  // workspace layout
  char* ws = (char*)d_ws;
  size_t off = 0;
  auto alloc = [&](size_t bytes) -> void* {
    void* p = ws + off;
    off = (off + bytes + 255) & ~(size_t)255;
    return p;
  };
  float*     x32    = (float*)    alloc((size_t)M2 * HID * 4);
  _Float16*  x16    = (_Float16*) alloc((size_t)M2 * HID * 2);
  _Float16*  qkv16  = (_Float16*) alloc((size_t)M2 * 192 * 2);
  _Float16*  attn16 = (_Float16*) alloc((size_t)M2 * HID * 2);
  float*     y32    = (float*)    alloc((size_t)M2 * HID * 4);
  _Float16*  ff116  = (_Float16*) alloc((size_t)M2 * 256 * 2);

  const int lnGrid = (M2 + 255) / 256;

  // Stage 1: per-(batch,band) GAT -> raw output in x32 (stage-2 seq layout)
  stage1_gat<<<dim3(B_SZ * NBANDS), 256, 0, stream>>>(
      nf, adj, iW, iAs, iAd, iWe, iAe, iBias, x32);
  // fused intra-LN + trans-LN
  ln_rows<<<lnGrid, 256, 0, stream>>>(x32, nullptr, iLng, iLnb, tLng, tLnb, x32, x16, M2);

  // Stage 2: transformer layers
  for (int l = 0; l < 2; ++l) {
    gemm_wmma<12><<<M2 / 128, 256, 0, stream>>>(
        x16, HID, qkvW + (size_t)l * 192 * HID, qkvB + (size_t)l * 192, HID,
        nullptr, qkv16, 0);
    attn_flash<<<B_SZ * HEADS * 4, 256, 0, stream>>>(qkv16, attn16);
    gemm_wmma<4><<<M2 / 128, 256, 0, stream>>>(
        attn16, HID, outW + (size_t)l * HID * HID, outB + (size_t)l * HID, HID,
        y32, nullptr, 0);
    ln_rows<<<lnGrid, 256, 0, stream>>>(x32, y32, ln1g + l * HID, ln1b + l * HID,
                                        nullptr, nullptr, x32, x16, M2);
    gemm_wmma<16><<<M2 / 128, 256, 0, stream>>>(
        x16, HID, ff1W + (size_t)l * 256 * HID, ff1B + (size_t)l * 256, HID,
        nullptr, ff116, 1);
    gemm_wmma<4><<<M2 / 128, 256, 0, stream>>>(
        ff116, 256, ff2W + (size_t)l * HID * 256, ff2B + (size_t)l * HID, 256,
        y32, nullptr, 0);
    ln_rows<<<lnGrid, 256, 0, stream>>>(x32, y32, ln2g + l * HID, ln2b + l * HID,
                                        nullptr, nullptr, x32, x16, M2);
  }

  // Stage 3: inter-band GAT + fusion attention + final LNs -> d_out [B,116,64]
  stage3_fuse<<<(B_SZ * NROIS) / 16, 256, 0, stream>>>(
      x16, interW, inAs, inAd, inWe, inAe, inBias,
      fQkvW, fQkvB, fOutW, fOutB, fusg, fusb, ilng, ilnb, (float*)d_out);
}
